// WindowAttention_24249385353297
// MI455X (gfx1250) — matrix operand order repaired and checked by reading, not compiled, not run on hardware
//
#include <hip/hip_runtime.h>

typedef __attribute__((ext_vector_type(8)))  _Float16 v8h;
typedef __attribute__((ext_vector_type(16))) _Float16 v16h;
typedef __attribute__((ext_vector_type(8)))  float    v8f;

// LDS layout strides (padded for bank-conflict avoidance)
#define XT_S 136   // token-major X: 64 tokens x 128 ch (halves)
#define XC_S 72    // channel-major X: 128 ch x 64 tokens (halves)
#define AF_S 68    // attn fp32: 64 x 64 (floats)
#define AH_S 72    // attn f16: 64 x 64 (halves)
#define OUT_S 133  // output staging: 64 tokens x 128 ch (floats)

#define XT_OFF 0
#define XC_OFF (XT_OFF + 64 * XT_S * 2)     // 17408
#define AF_OFF (XC_OFF + 128 * XC_S * 2)    // 35840
#define AH_OFF (AF_OFF + 64 * AF_S * 4)     // 53248
#define SMEM_BYTES (AH_OFF + 64 * AH_S * 2) // 62464

__device__ __forceinline__ v16h mk16(v8h lo, v8h hi) {
  union { v16h v; v8h h[2]; } u;
  u.h[0] = lo; u.h[1] = hi;
  return u.v;
}

__global__ __launch_bounds__(256) void win_attn_kernel(const float* __restrict__ x,
                                                       float* __restrict__ out) {
  __shared__ __align__(32) char smem[SMEM_BYTES];
  _Float16* Xt = (_Float16*)(smem + XT_OFF);
  _Float16* Xc = (_Float16*)(smem + XC_OFF);
  float*    Af = (float*)(smem + AF_OFF);
  _Float16* Ah = (_Float16*)(smem + AH_OFF);
  float*    Os = (float*)(smem + XT_OFF);   // overlays Xt+Xc after final barrier

  const int tid  = threadIdx.x;
  const int widx = blockIdx.x;              // 32768 windows
  const int b  = widx >> 12;
  const int wh = (widx >> 6) & 63;
  const int ww = widx & 63;
  const int base = ((b * 128) * 512 + wh * 8) * 512 + ww * 8;

  // ---- load window fp32 -> f16 into two LDS layouts ----
  #pragma unroll
  for (int it = 0; it < 4; ++it) {
    const int row = it * 256 + tid;         // row = c*8 + ty
    const int c  = row >> 3;
    const int ty = row & 7;
    const float4* p = (const float4*)(x + base + c * 262144 + ty * 512);
    float4 f0 = p[0];
    float4 f1 = p[1];
    float f[8] = {f0.x, f0.y, f0.z, f0.w, f1.x, f1.y, f1.z, f1.w};
    v8h h;
    #pragma unroll
    for (int tx = 0; tx < 8; ++tx) {
      _Float16 hv = (_Float16)f[tx];
      h[tx] = hv;
      Xt[(ty * 8 + tx) * XT_S + c] = hv;    // token-major
    }
    *(v8h*)&Xc[c * XC_S + ty * 8] = h;      // channel-major (contiguous 16B)
  }
  __syncthreads();

  const int wave = tid >> 5;
  const int lane = tid & 31;
  const int m  = lane & 15;                 // row (A) / column (B)
  const int hs = lane >> 4;                 // K-half selector

  // ---- QK^T: attn = scale * X X^T ----
  // Wave w owns row-block ti = w>>1 and column-blocks tj0, tj0+1 -> shared A.
  const float scale = 0.08838834764831845f; // 1/sqrt(128)
  {
    const int ti  = wave >> 1;
    const int tj0 = (wave & 1) * 2;
    v8f acc[2] = {{}, {}};
    #pragma unroll
    for (int kb = 0; kb < 4; ++kb) {
      const int k0 = kb * 32;
      v8h a_lo = *(const v8h*)&Xt[(ti * 16 + m) * XT_S + k0 + hs * 8];
      v8h a_hi = *(const v8h*)&Xt[(ti * 16 + m) * XT_S + k0 + 16 + hs * 8];
      const v16h a = mk16(a_lo, a_hi);
      #pragma unroll
      for (int t = 0; t < 2; ++t) {
        const int tj = tj0 + t;
        v8h b_lo = *(const v8h*)&Xt[(tj * 16 + m) * XT_S + k0 + hs * 8];
        v8h b_hi = *(const v8h*)&Xt[(tj * 16 + m) * XT_S + k0 + 16 + hs * 8];
        acc[t] = __builtin_amdgcn_wmma_f32_16x16x32_f16(false, a,
                                                        false, mk16(b_lo, b_hi),
                                                        (short)0, acc[t],
                                                        false, false);
      }
    }
    #pragma unroll
    for (int t = 0; t < 2; ++t) {
      const int tj = tj0 + t;
      #pragma unroll
      for (int r = 0; r < 8; ++r)
        Af[(ti * 16 + hs * 8 + r) * AF_S + tj * 16 + m] = acc[t][r] * scale;
    }
  }
  __syncthreads();

  // ---- softmax over 64 columns, 4 threads per row ----
  {
    const int row = tid >> 2;
    const int seg = tid & 3;
    const float4* pr = (const float4*)&Af[row * AF_S + seg * 16];
    float v[16];
    #pragma unroll
    for (int q = 0; q < 4; ++q) {
      float4 f = pr[q];
      v[q*4+0] = f.x; v[q*4+1] = f.y; v[q*4+2] = f.z; v[q*4+3] = f.w;
    }
    float mx = v[0];
    #pragma unroll
    for (int q = 1; q < 16; ++q) mx = fmaxf(mx, v[q]);
    mx = fmaxf(mx, __shfl_xor(mx, 1));
    mx = fmaxf(mx, __shfl_xor(mx, 2));
    float s = 0.0f;
    #pragma unroll
    for (int q = 0; q < 16; ++q) { v[q] = __expf(v[q] - mx); s += v[q]; }
    s += __shfl_xor(s, 1);
    s += __shfl_xor(s, 2);
    const float inv = 1.0f / s;
    v8h h0, h1;
    #pragma unroll
    for (int q = 0; q < 8; ++q) h0[q] = (_Float16)(v[q] * inv);
    #pragma unroll
    for (int q = 0; q < 8; ++q) h1[q] = (_Float16)(v[8 + q] * inv);
    *(v8h*)&Ah[row * AH_S + seg * 16]     = h0;
    *(v8h*)&Ah[row * AH_S + seg * 16 + 8] = h1;
  }
  __syncthreads();

  // ---- AV: out = attn * X ----
  // Wave w owns token-block ti = w>>1 and channel-blocks tn0..tn0+3 -> shared A.
  v8f oacc[4];
  const int av_ti  = wave >> 1;
  const int av_tn0 = (wave & 1) * 4;
  {
    #pragma unroll
    for (int t = 0; t < 4; ++t) oacc[t] = (v8f){};
    #pragma unroll
    for (int kb = 0; kb < 2; ++kb) {
      const int k0 = kb * 32;
      v8h a_lo = *(const v8h*)&Ah[(av_ti * 16 + m) * AH_S + k0 + hs * 8];
      v8h a_hi = *(const v8h*)&Ah[(av_ti * 16 + m) * AH_S + k0 + 16 + hs * 8];
      const v16h a = mk16(a_lo, a_hi);
      #pragma unroll
      for (int t = 0; t < 4; ++t) {
        const int tn = av_tn0 + t;
        v8h b_lo = *(const v8h*)&Xc[(tn * 16 + m) * XC_S + k0 + hs * 8];
        v8h b_hi = *(const v8h*)&Xc[(tn * 16 + m) * XC_S + k0 + 16 + hs * 8];
        oacc[t] = __builtin_amdgcn_wmma_f32_16x16x32_f16(false, a,
                                                         false, mk16(b_lo, b_hi),
                                                         (short)0, oacc[t],
                                                         false, false);
      }
    }
  }
  __syncthreads();                          // all Xc/Ah reads done; reuse LDS
  #pragma unroll
  for (int t = 0; t < 4; ++t) {
    const int tn = av_tn0 + t;
    #pragma unroll
    for (int r = 0; r < 8; ++r)
      Os[(av_ti * 16 + hs * 8 + r) * OUT_S + tn * 16 + m] = oacc[t][r];
  }
  __syncthreads();

  // ---- coalesced fp32 store (mirrors load pattern) ----
  #pragma unroll
  for (int it = 0; it < 4; ++it) {
    const int row = it * 256 + tid;
    const int c  = row >> 3;
    const int ty = row & 7;
    float o[8];
    #pragma unroll
    for (int tx = 0; tx < 8; ++tx) o[tx] = Os[(ty * 8 + tx) * OUT_S + c];
    float4* po = (float4*)(out + base + c * 262144 + ty * 512);
    po[0] = make_float4(o[0], o[1], o[2], o[3]);
    po[1] = make_float4(o[4], o[5], o[6], o[7]);
  }
}

extern "C" void kernel_launch(void* const* d_in, const int* in_sizes, int n_in,
                              void* d_out, int out_size, void* d_ws, size_t ws_size,
                              hipStream_t stream) {
  (void)in_sizes; (void)n_in; (void)out_size; (void)d_ws; (void)ws_size;
  const float* x = (const float*)d_in[0];
  float* out = (float*)d_out;
  dim3 grid(32768);   // 8 * 64 * 64 windows
  dim3 block(256);    // 8 waves (wave32)
  hipLaunchKernelGGL(win_attn_kernel, grid, block, 0, stream, x, out);
}
